// Model_26594437497231
// MI455X (gfx1250) — matrix operand order repaired and checked by reading, not compiled, not run on hardware
//
#include <hip/hip_runtime.h>
#include <hip/hip_fp16.h>

typedef __attribute__((ext_vector_type(16))) _Float16 v16h;
typedef __attribute__((ext_vector_type(8)))  float    v8f;
typedef unsigned int  tdm_u32x4 __attribute__((ext_vector_type(4)));
typedef int           tdm_i32x8 __attribute__((ext_vector_type(8)));
typedef int           tdm_i32x4 __attribute__((ext_vector_type(4)));

#define S_LEN 8192
#define D_DIM 300
#define U_DIM 150
#define KT    10     // K tiles of 32 (300 -> 320)
#define MT    512    // M tiles of 16 (8192/16)
#define NT1   76     // N tiles for x@[kf|kb]  (1200 -> 1216)
#define LD1   1216
#define NT2   20     // N tiles for hidden@W1 (300 -> 320, NB=4 friendly)
#define LD2   320
#define RKW   45000  // packed rk dwords per direction (600 cols * 75 pairs)

// ---------------------------------------------------------------------------
// A-matrix (16x32 f16) lane/element -> K mapping used here: K = 16*(lane/16) + e (contiguous, same as B)
// ---------------------------------------------------------------------------
__device__ __forceinline__ int a_kk_of(int lane, int e) {
    int half = lane >> 4;
    int v = e >> 1, p = e & 1;
    return 16 * half + 2 * v + p;
}

// Pack A tiles directly from the embedding gather (fuses gather + f16 convert).
__global__ void pack_a_gather(const int* __restrict__ sentence,
                              const float* __restrict__ E,
                              _Float16* __restrict__ dst) {
    int gid = blockIdx.x * blockDim.x + threadIdx.x;
    int total = MT * KT * 32 * 16;
    if (gid >= total) return;
    int e = gid & 15;
    int lane = (gid >> 4) & 31;
    int t = gid >> 9;
    int kt = t % KT, mt = t / KT;
    int m = mt * 16 + (lane & 15);
    int k = kt * 32 + a_kk_of(lane, e);
    float v = 0.f;
    if (k < D_DIM) {
        int word = sentence[m];
        v = E[(long long)word * D_DIM + k];
    }
    dst[gid] = (_Float16)v;
}

// Pack A tiles from a dense f32 matrix (hidden states).
__global__ void pack_a_dense(const float* __restrict__ src,
                             _Float16* __restrict__ dst) {
    int gid = blockIdx.x * blockDim.x + threadIdx.x;
    int total = MT * KT * 32 * 16;
    if (gid >= total) return;
    int e = gid & 15;
    int lane = (gid >> 4) & 31;
    int t = gid >> 9;
    int kt = t % KT, mt = t / KT;
    int m = mt * 16 + (lane & 15);
    int k = kt * 32 + a_kk_of(lane, e);
    float v = (k < D_DIM) ? src[m * D_DIM + k] : 0.f;
    dst[gid] = (_Float16)v;
}

// B-matrix (32x16 f16): lane n -> N = n%16; K = (n/16)*16 + e.
// Combined [kf | kb] logical matrix, 300 x 1200.
__global__ void pack_b_lstm(const float* __restrict__ kf,
                            const float* __restrict__ kb,
                            _Float16* __restrict__ dst) {
    int gid = blockIdx.x * blockDim.x + threadIdx.x;
    int total = KT * NT1 * 32 * 16;
    if (gid >= total) return;
    int e = gid & 15;
    int lane = (gid >> 4) & 31;
    int t = gid >> 9;
    int kt = t % KT, nt = t / KT;
    int n = nt * 16 + (lane & 15);
    int k = kt * 32 + (lane >> 4) * 16 + e;
    float v = 0.f;
    if (k < D_DIM && n < 1200)
        v = (n < 600) ? kf[k * 600 + n] : kb[k * 600 + (n - 600)];
    dst[gid] = (_Float16)v;
}

// Generic B pack (W1: 300 x 300, ld 300).
__global__ void pack_b_generic(const float* __restrict__ src,
                               _Float16* __restrict__ dst,
                               int ntiles, int kreal, int nreal, int ld) {
    int gid = blockIdx.x * blockDim.x + threadIdx.x;
    int total = KT * ntiles * 32 * 16;
    if (gid >= total) return;
    int e = gid & 15;
    int lane = (gid >> 4) & 31;
    int t = gid >> 9;
    int kt = t % KT, nt = t / KT;
    int n = nt * 16 + (lane & 15);
    int k = kt * 32 + (lane >> 4) * 16 + e;
    float v = (k < kreal && n < nreal) ? src[k * ld + n] : 0.f;
    dst[gid] = (_Float16)v;
}

// rk (150x600 f32) -> packed f16 stream in the EXACT LDS layout the scan uses
// (column-major __half2 pairs, stage[dir][j*75 + i2]).  The TDM then moves the
// whole 180 KB block global->LDS as one 1-D tile.
__global__ void pack_rk(const float* __restrict__ rkf,
                        const float* __restrict__ rkb,
                        __half2* __restrict__ stage) {
    int gid = blockIdx.x * blockDim.x + threadIdx.x;
    if (gid >= 2 * RKW) return;
    int dir = gid / RKW, d = gid % RKW;
    const float* rk = dir ? rkb : rkf;
    int j = d / 75, i2 = d % 75;
    __half2 w;
    w.x = __float2half(rk[(2 * i2) * 600 + j]);
    w.y = __float2half(rk[(2 * i2 + 1) * 600 + j]);
    stage[gid] = w;
}

// Bias padding + H zeroing (H must be re-zeroed every launch).
__global__ void init_misc(const float* __restrict__ bf, const float* __restrict__ bb,
                          const float* __restrict__ b1,
                          float* __restrict__ bias1, float* __restrict__ bias2,
                          float* __restrict__ H) {
    int i = blockIdx.x * blockDim.x + threadIdx.x;
    if (i < LD1)
        bias1[i] = (i < 600) ? bf[i] : (i < 1200 ? bb[i - 600] : 0.f);
    else if (i < LD1 + LD2) {
        int j = i - LD1;
        bias2[j] = (j < D_DIM) ? b1[j] : 0.f;
    } else if (i < LD1 + 2 * LD2) {
        H[i - LD1 - LD2] = 0.f;
    }
}

// ---------------------------------------------------------------------------
// WMMA GEMM, register-blocked: each wave owns a 2x4 block of 16x16 tiles.
// Per k-step: 2 A loads + 4 B loads -> 8 WMMAs (A reused x4, B reused x2).
// ---------------------------------------------------------------------------
__global__ void wmma_gemm(const _Float16* __restrict__ Ap,
                          const _Float16* __restrict__ Bp,
                          const float* __restrict__ bias,
                          float* __restrict__ C,
                          int ntiles, int ldc) {
    int lane = threadIdx.x & 31;
    int ngroups = ntiles >> 2;                      // NB = 4
    int grp = blockIdx.x * 8 + (threadIdx.x >> 5);
    if (grp >= (MT >> 1) * ngroups) return;         // MB = 2
    int mg = grp / ngroups, ng = grp % ngroups;

    const _Float16* ap0 = Ap + (((long long)(mg * 2) * KT) * 32 + lane) * 16;
    const _Float16* ap1 = ap0 + (long long)KT * 512;

    v8f acc[2][4] = {};
#pragma unroll
    for (int kt = 0; kt < KT; ++kt) {
        v16h a0 = *(const v16h*)(ap0 + (long long)kt * 512);
        v16h a1 = *(const v16h*)(ap1 + (long long)kt * 512);
#pragma unroll
        for (int n = 0; n < 4; ++n) {
            v16h b = *(const v16h*)(Bp +
                (((long long)(ng * 4 + n) * KT + kt) * 32 + lane) * 16);
            acc[0][n] = __builtin_amdgcn_wmma_f32_16x16x32_f16(
                false, a0, false, b, (short)0, acc[0][n], false, false);
            acc[1][n] = __builtin_amdgcn_wmma_f32_16x16x32_f16(
                false, a1, false, b, (short)0, acc[1][n], false, false);
        }
    }
    // C/D layout: lane L holds col N = L%16; vgpr r -> row r + 8*(L/16)
#pragma unroll
    for (int m = 0; m < 2; ++m) {
        int rb = (mg * 2 + m) * 16 + (lane >> 4) * 8;
#pragma unroll
        for (int n = 0; n < 4; ++n) {
            int col = (ng * 4 + n) * 16 + (lane & 15);
            float bv = bias[col];
#pragma unroll
            for (int r = 0; r < 8; ++r)
                C[(long long)(rb + r) * ldc + col] = acc[m][n][r] + bv;
        }
    }
}

// ---------------------------------------------------------------------------
// Bi-LSTM scan: block 0 = forward, block 1 = backward. 640 threads.
// rk resident in LDS as packed f16 (180 KB of the WGP's 320 KB), delivered by
// the Tensor Data Mover as a single 1-D tile (CDNA5 ISA).
// ---------------------------------------------------------------------------
__global__ void lstm_scan(const __half2* __restrict__ rkstage,  // [2][45000]
                          const float* __restrict__ X,          // [S][1216]
                          float* __restrict__ hidden) {         // [S][300]
    extern __shared__ char smem_raw[];
    __half2* rk2 = (__half2*)smem_raw;                 // 45000 dwords (180000 B)
    __half2* hpk = (__half2*)(smem_raw + 180224);      // 76 dwords packed h (f16)
    float*   z   = (float*)  (smem_raw + 180736);      // 600 f32 gate pre-acts

    const int dir = blockIdx.x;
    const int tid = threadIdx.x;

    if (tid < 76) ((unsigned int*)hpk)[tid] = 0u;      // h0 = 0

#if __has_builtin(__builtin_amdgcn_tensor_load_to_lds)
    if (tid == 0) {
        unsigned long long ga =
            (unsigned long long)(const void*)(rkstage + (long long)dir * RKW);
        unsigned int lds_off = (unsigned int)(unsigned long long)(void*)smem_raw;
        // D# group 0: count=1 | lds_addr | global_addr[56:0] | type=2
        tdm_u32x4 g0;
        g0[0] = 1u;
        g0[1] = lds_off;
        g0[2] = (unsigned int)(ga & 0xFFFFFFFFu);
        g0[3] = (unsigned int)((ga >> 32) & 0x01FFFFFFu) | (2u << 30);
        // D# group 1: data_size=4B; tensor_dim0=tile_dim0=stride0=45000 dwords;
        // tensor_dim1=tile_dim1=1 (1-D copy)
        tdm_i32x8 g1;
        g1[0] = 0x00020000;                  // data_size=2 (4 bytes)
        g1[1] = (int)0xAFC80000;             // tensor_dim0[15:0]=45000 << 16
        g1[2] = 0x00010000;                  // tensor_dim0 hi | tensor_dim1=1
        g1[3] = (int)0xAFC80000;             // tile_dim0=45000 << 16
        g1[4] = 1;                           // tile_dim1=1
        g1[5] = RKW;                         // tensor_dim0_stride
        g1[6] = 0;
        g1[7] = 0;
        tdm_i32x4 z4 = {0, 0, 0, 0};
  #if __has_include(<hip/amd_detail/amd_gfx1250_TDM.h>)
        tdm_i32x8 z8 = {0, 0, 0, 0, 0, 0, 0, 0};
        __builtin_amdgcn_tensor_load_to_lds(g0, g1, z4, z4, z8, 0);
  #else
        __builtin_amdgcn_tensor_load_to_lds(g0, g1, z4, z4, 0);
  #endif
        __builtin_amdgcn_s_wait_tensorcnt(0);
    }
#else
    for (int d = tid; d < RKW; d += 640)
        rk2[d] = rkstage[(long long)dir * RKW + d];
#endif
    __syncthreads();

    const __half2* col = rk2 + tid * 75;
    float c = 0.f;
    __half* hph = (__half*)hpk;

    for (int step = 0; step < S_LEN; ++step) {
        int t = dir ? (S_LEN - 1 - step) : step;
        if (tid < 600) {
            float acc = X[(long long)t * LD1 + dir * 600 + tid];
#pragma unroll 5
            for (int i2 = 0; i2 < 75; ++i2) {
                __half2 w = col[i2];
                __half2 h2 = hpk[i2];
                acc += __half2float(w.x) * __half2float(h2.x)
                     + __half2float(w.y) * __half2float(h2.y);
            }
            z[tid] = acc;
        }
        __syncthreads();                                // z ready; hpk reads done
        if (tid < U_DIM) {
            float zi = z[tid], zf = z[150 + tid], zg = z[300 + tid], zo = z[450 + tid];
            float ig = 1.f / (1.f + __expf(-zi));
            float fg = 1.f / (1.f + __expf(-zf));
            float og = 1.f / (1.f + __expf(-zo));
            c = fg * c + ig * tanhf(zg);
            float h = og * tanhf(c);
            hidden[(long long)t * D_DIM + dir * U_DIM + tid] = h;
            hph[tid] = __float2half(h);
        }
        __syncthreads();                                // hpk ready for next step
    }
}

// ---------------------------------------------------------------------------
// Primary + secondary attention: one wave per sentence (x16), shfl reductions,
// per-lane partial H flushed once via f32 atomics.
// ---------------------------------------------------------------------------
__device__ __forceinline__ float wave_sum(float v) {
#pragma unroll
    for (int m = 16; m >= 1; m >>= 1) v += __shfl_xor(v, m, 32);
    return v;
}

__global__ void attention(const int* __restrict__ sentence,
                          const int* __restrict__ synidx,
                          const float* __restrict__ E,
                          const float* __restrict__ outm,   // [S][LD2]
                          const float* __restrict__ hidden, // [S][300]
                          const float* __restrict__ W2,
                          const float* __restrict__ b2,
                          float* __restrict__ H) {          // [LD2] accum
    int lane = threadIdx.x & 31;
    int gw = blockIdx.x * 8 + (threadIdx.x >> 5);           // 512 waves total
    float Hacc[10];
#pragma unroll
    for (int r = 0; r < 10; ++r) Hacc[r] = 0.f;

    for (int si = 0; si < 16; ++si) {
        int s = gw * 16 + si;
        float o_r[10], h_r[10], sv[4][10];
        int word = sentence[s];
#pragma unroll
        for (int r = 0; r < 10; ++r) {
            int d = lane + 32 * r;
            bool ok = d < D_DIM;
            o_r[r] = ok ? outm[(long long)s * LD2 + d] : 0.f;
            h_r[r] = ok ? hidden[(long long)s * D_DIM + d] : 0.f;
        }
        float coeff[4];
#pragma unroll
        for (int k = 0; k < 4; ++k) {
            int idx = synidx[word * 4 + k];
            const float* row = E + (long long)idx * D_DIM;
            __builtin_prefetch(row, 0, 0);
            float dot = 0.f;
#pragma unroll
            for (int r = 0; r < 10; ++r) {
                int d = lane + 32 * r;
                float v = (d < D_DIM) ? row[d] : 0.f;
                sv[k][r] = v;
                dot += v * o_r[r];
            }
            coeff[k] = __expf(wave_sum(dot));
        }
        float dot2 = 0.f;
        float hh_r[10];
#pragma unroll
        for (int r = 0; r < 10; ++r) {
            float m = coeff[0] * sv[0][r] + coeff[1] * sv[1][r]
                    + coeff[2] * sv[2][r] + coeff[3] * sv[3][r];
            float hh = m + h_r[r];
            hh_r[r] = hh;
            int d = lane + 32 * r;
            dot2 += hh * ((d < D_DIM) ? W2[d] : 0.f);
        }
        float c2 = __expf(tanhf(wave_sum(dot2) + b2[0]));
#pragma unroll
        for (int r = 0; r < 10; ++r) Hacc[r] += c2 * hh_r[r];
    }
#pragma unroll
    for (int r = 0; r < 10; ++r) {
        int d = lane + 32 * r;
        if (d < D_DIM) atomicAdd(&H[d], Hacc[r]);
    }
}

// Final heads: 9 outputs (8 emotion {0,1}, 1 sentiment {+1,-1}).
__global__ void heads(const float* __restrict__ H,
                      const float* __restrict__ We, const float* __restrict__ be,
                      const float* __restrict__ Ws, const float* __restrict__ bs,
                      float* __restrict__ out) {
    int j = threadIdx.x;
    if (j >= 9) return;
    float acc = (j < 8) ? be[j] : bs[0];
    for (int d = 0; d < D_DIM; ++d)
        acc += H[d] * ((j < 8) ? We[d * 8 + j] : Ws[d]);
    float p = 1.f / (1.f + __expf(-acc));
    out[j] = (j < 8) ? (p > 0.5f ? 1.f : 0.f) : (p > 0.5f ? 1.f : -1.f);
}

// ---------------------------------------------------------------------------
extern "C" void kernel_launch(void* const* d_in, const int* in_sizes, int n_in,
                              void* d_out, int out_size, void* d_ws, size_t ws_size,
                              hipStream_t stream) {
    const int*   sentence = (const int*)  d_in[0];
    const float* E        = (const float*)d_in[1];
    const int*   synidx   = (const int*)  d_in[2];
    const float* kf       = (const float*)d_in[3];
    const float* rkf      = (const float*)d_in[4];
    const float* bf       = (const float*)d_in[5];
    const float* kb       = (const float*)d_in[6];
    const float* rkb      = (const float*)d_in[7];
    const float* bb       = (const float*)d_in[8];
    const float* W1       = (const float*)d_in[9];
    const float* b1       = (const float*)d_in[10];
    const float* W2       = (const float*)d_in[11];
    const float* b2       = (const float*)d_in[12];
    const float* We       = (const float*)d_in[13];
    const float* be       = (const float*)d_in[14];
    const float* Ws       = (const float*)d_in[15];
    const float* bs       = (const float*)d_in[16];
    float* out = (float*)d_out;

    // Workspace carve-up (256B aligned regions)
    char* ws = (char*)d_ws;
    size_t off = 0;
    auto carve = [&](size_t bytes) { char* p = ws + off;
        off = (off + bytes + 255) & ~(size_t)255; return p; };
    _Float16* Ap1   = (_Float16*)carve((size_t)MT * KT * 512 * 2);   // 5.24 MB
    _Float16* Bp1   = (_Float16*)carve((size_t)KT * NT1 * 512 * 2);  // 0.78 MB
    float*    X     = (float*)   carve((size_t)S_LEN * LD1 * 4);     // 39.8 MB
    float*    bias1 = (float*)   carve((size_t)LD1 * 4);
    float*    hid   = (float*)   carve((size_t)S_LEN * D_DIM * 4);   // 9.8 MB
    _Float16* Ap2   = (_Float16*)carve((size_t)MT * KT * 512 * 2);   // 5.24 MB
    _Float16* Bp2   = (_Float16*)carve((size_t)KT * NT2 * 512 * 2);
    float*    bias2 = (float*)   carve((size_t)LD2 * 4);
    float*    outm  = (float*)   carve((size_t)S_LEN * LD2 * 4);     // 10.5 MB
    float*    H     = (float*)   carve((size_t)LD2 * 4);
    __half2*  rkst  = (__half2*) carve((size_t)2 * RKW * 4);         // 0.36 MB
    (void)ws_size; (void)in_sizes; (void)n_in; (void)out_size;

    const int packA  = MT * KT * 512;      // 2,621,440
    const int packB1 = KT * NT1 * 512;     // 389,120
    const int packB2 = KT * NT2 * 512;     // 102,400

    // 1) gather+pack embeddings, pack weights, stage rk in LDS layout, init misc
    pack_a_gather<<<(packA + 255) / 256, 256, 0, stream>>>(sentence, E, Ap1);
    pack_b_lstm<<<(packB1 + 255) / 256, 256, 0, stream>>>(kf, kb, Bp1);
    pack_rk<<<(2 * RKW + 255) / 256, 256, 0, stream>>>(rkf, rkb, rkst);
    init_misc<<<(LD1 + 2 * LD2 + 255) / 256, 256, 0, stream>>>(bf, bb, b1, bias1, bias2, H);

    // 2) X = emb @ [kf|kb] + [bf|bb]   (hoisted out of the recurrence)
    wmma_gemm<<<(MT / 2) * (NT1 / 4) / 8, 256, 0, stream>>>(Ap1, Bp1, bias1, X, NT1, LD1);

    // 3) sequential bi-LSTM scan (fwd/bwd in parallel workgroups; rk via TDM)
    lstm_scan<<<2, 640, 183360, stream>>>(rkst, X, hid);

    // 4) out = hidden @ W1 + b1
    pack_a_dense<<<(packA + 255) / 256, 256, 0, stream>>>(hid, Ap2);
    pack_b_generic<<<(packB2 + 255) / 256, 256, 0, stream>>>(W1, Bp2, NT2, D_DIM, D_DIM, D_DIM);
    wmma_gemm<<<(MT / 2) * (NT2 / 4) / 8, 256, 0, stream>>>(Ap2, Bp2, bias2, outm, NT2, LD2);

    // 5) attention (primary synonym + secondary) -> H (atomically accumulated)
    attention<<<64, 256, 0, stream>>>(sentence, synidx, E, outm, hid, W2, b2, H);

    // 6) heads -> 9 outputs
    heads<<<1, 32, 0, stream>>>(H, We, be, Ws, bs, out);
}
